// Multihead_45612552683619
// MI455X (gfx1250) — matrix operand order repaired and checked by reading, not compiled, not run on hardware
//
#include <hip/hip_runtime.h>
#include <hip/hip_bf16.h>

#define D_FEAT 1024
#define NHEAD  16
#define DKV    64
#define SEQ    2048
#define BATCH  2
#define MROWS  (BATCH * SEQ)   // 4096

typedef __bf16 bf16_t;
typedef __attribute__((ext_vector_type(16))) __bf16 v16bf;
typedef __attribute__((ext_vector_type(8)))  float  v8f;
typedef __attribute__((ext_vector_type(4)))  unsigned int u32x4;
typedef __attribute__((ext_vector_type(4)))  float  f32x4;

union Frag16 {
    v16bf v;
    unsigned short us[16];
    u32x4 q[2];
};

__device__ __forceinline__ unsigned short f2bf(float f) {
    unsigned u = __builtin_bit_cast(unsigned, f);
    u += 0x7FFFu + ((u >> 16) & 1u);          // round-to-nearest-even
    return (unsigned short)(u >> 16);
}

__device__ __forceinline__ v8f wmma_bf16(const Frag16& a, const Frag16& b, v8f c) {
    return __builtin_amdgcn_wmma_f32_16x16x32_bf16(false, a.v, false, b.v,
                                                   (short)0, c, false, false);
}

// ---------------------------------------------------------------------------
// Elementwise f32 -> bf16 (8 elems/thread, b128 in / b128 out)
// ---------------------------------------------------------------------------
__global__ __launch_bounds__(256) void cvt_bf16_kernel(
    const float* __restrict__ src, unsigned short* __restrict__ dst, int n8) {
    int i = blockIdx.x * 256 + threadIdx.x;
    if (i >= n8) return;
    const f32x4* s = (const f32x4*)(src + (size_t)i * 8);
    f32x4 a = s[0], b = s[1];
    u32x4 o;
    o[0] = (unsigned)f2bf(a[0]) | ((unsigned)f2bf(a[1]) << 16);
    o[1] = (unsigned)f2bf(a[2]) | ((unsigned)f2bf(a[3]) << 16);
    o[2] = (unsigned)f2bf(b[0]) | ((unsigned)f2bf(b[1]) << 16);
    o[3] = (unsigned)f2bf(b[2]) | ((unsigned)f2bf(b[3]) << 16);
    *(u32x4*)(dst + (size_t)i * 8) = o;
}

// ---------------------------------------------------------------------------
// Weight convert+transpose: Wt[n][k] = bf16(W[k][n]),  K x N fp32 -> N x K bf16
// ---------------------------------------------------------------------------
__global__ __launch_bounds__(256) void transpose_cvt_kernel(
    const float* __restrict__ W, unsigned short* __restrict__ Wt, int K, int N) {
    int idx = blockIdx.x * 256 + threadIdx.x;
    if (idx >= K * N) return;
    int k = idx % K, n = idx / K;
    Wt[idx] = f2bf(W[(size_t)k * N + n]);
}

// ---------------------------------------------------------------------------
// Unified GEMM: acc = actb(bf16)[M,1024] @ wt^T(bf16) + bias
//   mode 0: dstb[((b*H+h)*S+s)*64+d]          (Q, K layout)
//   mode 1: dstb[((b*H+h)*64+d)*S+s]          (V transposed layout)
//   mode 2: dstf[m*1024+n] = acc + resid[...] (f32 + residual)
// grid (M/128, 1024/64), block 128 (4 waves); wave computes 32x64
// (two A-fragments share every B-fragment -> 8 WMMAs per k-step).
// ---------------------------------------------------------------------------
__global__ __launch_bounds__(128) void gemm_kernel(
    const unsigned short* __restrict__ actb,
    const unsigned short* __restrict__ wt,
    const float* __restrict__ bias,
    unsigned short* __restrict__ dstb,
    const float* __restrict__ resid,
    float* __restrict__ dstf,
    int mode) {
    const int wave = threadIdx.x >> 5, lane = threadIdx.x & 31;
    const int l16 = lane & 15, kh = lane >> 4, base = kh * 8;
    const int m0 = blockIdx.x * 128 + wave * 32;
    const int n0 = blockIdx.y * 64;

    v8f acc[2][4];
#pragma unroll
    for (int nt = 0; nt < 4; ++nt) {
        float bv = bias[n0 + nt * 16 + l16];
#pragma unroll
        for (int half = 0; half < 2; ++half)
#pragma unroll
            for (int r = 0; r < 8; ++r) acc[half][nt][r] = bv;
    }

    const unsigned short* ap0 = actb + (size_t)(m0 + l16) * D_FEAT;
    const unsigned short* ap1 = ap0 + 16 * D_FEAT;
    for (int k0 = 0; k0 < D_FEAT; k0 += 32) {
        Frag16 A0, A1;
        A0.q[0] = *(const u32x4*)(ap0 + k0 + base);
        A0.q[1] = *(const u32x4*)(ap0 + k0 + base + 16);
        A1.q[0] = *(const u32x4*)(ap1 + k0 + base);
        A1.q[1] = *(const u32x4*)(ap1 + k0 + base + 16);
#pragma unroll
        for (int nt = 0; nt < 4; ++nt) {
            const unsigned short* wp =
                wt + (size_t)(n0 + nt * 16 + l16) * D_FEAT + k0 + kh * 8;
            Frag16 Bf;
            Bf.q[0] = *(const u32x4*)(wp);
            Bf.q[1] = *(const u32x4*)(wp + 16);
            acc[0][nt] = wmma_bf16(A0, Bf, acc[0][nt]);
            acc[1][nt] = wmma_bf16(A1, Bf, acc[1][nt]);
        }
    }

#pragma unroll
    for (int half = 0; half < 2; ++half)
#pragma unroll
        for (int nt = 0; nt < 4; ++nt)
#pragma unroll
            for (int r = 0; r < 8; ++r) {
                int m = m0 + half * 16 + kh * 8 + r;
                int n = n0 + nt * 16 + l16;
                float val = acc[half][nt][r];
                if (mode == 2) {
                    size_t idx = (size_t)m * D_FEAT + n;
                    dstf[idx] = val + resid[idx];
                } else {
                    int b = m >> 11, s = m & (SEQ - 1);
                    int h = n >> 6, d = n & 63;
                    unsigned short o = f2bf(val);
                    if (mode == 0)
                        dstb[(((size_t)(b * NHEAD + h) * SEQ + s) << 6) + d] = o;
                    else
                        dstb[((size_t)(b * NHEAD + h) * 64 + d) * SEQ + s] = o;
                }
            }
}

// ---------------------------------------------------------------------------
// Flash attention: one wave = 16 query rows of one (b,h). block = 4 waves.
// ---------------------------------------------------------------------------
__global__ __launch_bounds__(128) void attention_kernel(
    const unsigned short* __restrict__ qproj,
    const unsigned short* __restrict__ kproj,
    const unsigned short* __restrict__ vT,
    unsigned short* __restrict__ ctxbuf) {
    __shared__ __align__(16) unsigned short p_lds[4][16 * 32];
    const int wave = threadIdx.x >> 5, lane = threadIdx.x & 31;
    const int l16 = lane & 15, kh = lane >> 4, base = kh * 8;
    const int wid = blockIdx.x * 4 + wave;
    const int qt = wid & (SEQ / 16 - 1);
    const int h  = (wid >> 7) & (NHEAD - 1);
    const int b  = wid >> 11;
    const size_t bh = (size_t)(b * NHEAD + h);

    // Q A-fragments for d 0..31 and 32..63
    Frag16 Qf[2];
    const unsigned short* qp = qproj + (bh * SEQ + qt * 16 + l16) * 64;
#pragma unroll
    for (int f = 0; f < 2; ++f) {
        const unsigned short* p = qp + f * 32 + base;
        Qf[f].q[0] = *(const u32x4*)(p);
        Qf[f].q[1] = *(const u32x4*)(p + 16);
    }

    float m_run[8], l_run[8];
    v8f ctx[4];
#pragma unroll
    for (int r = 0; r < 8; ++r) { m_run[r] = -3.0e38f; l_run[r] = 0.f; }
#pragma unroll
    for (int nt = 0; nt < 4; ++nt)
#pragma unroll
        for (int r = 0; r < 8; ++r) ctx[nt][r] = 0.f;

    const unsigned short* kp = kproj + bh * SEQ * 64;
    const unsigned short* vp = vT + bh * 64 * SEQ;
    unsigned short* pl = p_lds[wave];

    for (int s0 = 0; s0 < SEQ; s0 += 32) {
        // scores for 2 tiles of 16 kv columns
        v8f sc[2];
#pragma unroll
        for (int t = 0; t < 2; ++t) {
            v8f a;
#pragma unroll
            for (int r = 0; r < 8; ++r) a[r] = 0.f;
#pragma unroll
            for (int kk = 0; kk < 2; ++kk) {
                const unsigned short* bp =
                    kp + (size_t)(s0 + t * 16 + l16) * 64 + kk * 32 + kh * 8;
                Frag16 Bf;
                Bf.q[0] = *(const u32x4*)bp;
                Bf.q[1] = *(const u32x4*)(bp + 16);
                a = wmma_bf16(Qf[kk], Bf, a);
            }
            sc[t] = a;
        }
        // online softmax over this 32-column chunk
#pragma unroll
        for (int r = 0; r < 8; ++r) {
            float v0 = sc[0][r] * 0.125f;   // 1/sqrt(64)
            float v1 = sc[1][r] * 0.125f;
            float mx = fmaxf(v0, v1);
#pragma unroll
            for (int msk = 1; msk < 16; msk <<= 1)
                mx = fmaxf(mx, __shfl_xor(mx, msk, 32));
            float mnew  = fmaxf(m_run[r], mx);
            float alpha = __expf(m_run[r] - mnew);
            float p0 = __expf(v0 - mnew), p1 = __expf(v1 - mnew);
            float ss = p0 + p1;
#pragma unroll
            for (int msk = 1; msk < 16; msk <<= 1)
                ss += __shfl_xor(ss, msk, 32);
            l_run[r] = l_run[r] * alpha + ss;
            m_run[r] = mnew;
#pragma unroll
            for (int nt = 0; nt < 4; ++nt) ctx[nt][r] *= alpha;
            int rw = kh * 8 + r;
            pl[rw * 32 + l16]      = f2bf(p0);
            pl[rw * 32 + 16 + l16] = f2bf(p1);
        }
        asm volatile("s_wait_dscnt 0" ::: "memory");
        // P A-fragment from LDS (layout swap D-frag -> A-frag)
        Frag16 Pf;
        {
            const unsigned short* pp = pl + l16 * 32 + base;
            Pf.q[0] = *(const u32x4*)pp;
            Pf.q[1] = *(const u32x4*)(pp + 16);
        }
        // ctx += P @ V  (V^T layout -> contiguous B fragments)
#pragma unroll
        for (int nt = 0; nt < 4; ++nt) {
            const unsigned short* bp =
                vp + (size_t)(nt * 16 + l16) * SEQ + s0 + kh * 8;
            Frag16 Bf;
            Bf.q[0] = *(const u32x4*)bp;
            Bf.q[1] = *(const u32x4*)(bp + 16);
            ctx[nt] = wmma_bf16(Pf, Bf, ctx[nt]);
        }
    }

#pragma unroll
    for (int nt = 0; nt < 4; ++nt)
#pragma unroll
        for (int r = 0; r < 8; ++r) {
            float val = ctx[nt][r] / l_run[r];
            int s = qt * 16 + kh * 8 + r;
            int col = h * 64 + nt * 16 + l16;
            ctxbuf[(size_t)(b * SEQ + s) * D_FEAT + col] = f2bf(val);
        }
}

// ---------------------------------------------------------------------------
// In-place row LayerNorm (1024 cols, 256 threads, 4 cols/thread)
// ---------------------------------------------------------------------------
__global__ __launch_bounds__(256) void layernorm_kernel(
    float* __restrict__ out, const float* __restrict__ gamma,
    const float* __restrict__ beta) {
    __shared__ float rs[8], rs2[8];
    const int t = threadIdx.x, wave = t >> 5, lane = t & 31;
    float* p = out + (size_t)blockIdx.x * D_FEAT;
    f32x4 x = *(const f32x4*)(p + t * 4);
    float s = x[0] + x[1] + x[2] + x[3];
    float s2 = x[0] * x[0] + x[1] * x[1] + x[2] * x[2] + x[3] * x[3];
#pragma unroll
    for (int m = 1; m < 32; m <<= 1) {
        s  += __shfl_xor(s, m, 32);
        s2 += __shfl_xor(s2, m, 32);
    }
    if (lane == 0) { rs[wave] = s; rs2[wave] = s2; }
    __syncthreads();
    float S = 0.f, S2 = 0.f;
#pragma unroll
    for (int i = 0; i < 8; ++i) { S += rs[i]; S2 += rs2[i]; }
    float mean = S * (1.0f / D_FEAT);
    float var  = S2 * (1.0f / D_FEAT) - mean * mean;
    float rstd = rsqrtf(var + 1e-6f);
    f32x4 g  = *(const f32x4*)(gamma + t * 4);
    f32x4 be = *(const f32x4*)(beta + t * 4);
    f32x4 y;
#pragma unroll
    for (int j = 0; j < 4; ++j) y[j] = (x[j] - mean) * rstd * g[j] + be[j];
    *(f32x4*)(p + t * 4) = y;
}

// ---------------------------------------------------------------------------
extern "C" void kernel_launch(void* const* d_in, const int* in_sizes, int n_in,
                              void* d_out, int out_size, void* d_ws, size_t ws_size,
                              hipStream_t stream) {
    const float* q     = (const float*)d_in[0];
    const float* k     = (const float*)d_in[1];
    const float* v     = (const float*)d_in[2];
    const float* Wq    = (const float*)d_in[3];
    const float* bq    = (const float*)d_in[4];
    const float* Wk    = (const float*)d_in[5];
    const float* bk    = (const float*)d_in[6];
    const float* Wv    = (const float*)d_in[7];
    const float* bv    = (const float*)d_in[8];
    const float* Wo    = (const float*)d_in[9];
    const float* bo    = (const float*)d_in[10];
    const float* gamma = (const float*)d_in[11];
    const float* beta  = (const float*)d_in[12];
    float* out = (float*)d_out;

    char* ws = (char*)d_ws;
    const size_t WBYTES = (size_t)D_FEAT * D_FEAT * 2;       // 2 MB
    const size_t PBYTES = (size_t)MROWS * D_FEAT * 2;        // 8 MB
    unsigned short* WqT   = (unsigned short*)(ws);             ws += WBYTES;
    unsigned short* WkT   = (unsigned short*)(ws);             ws += WBYTES;
    unsigned short* WvT   = (unsigned short*)(ws);             ws += WBYTES;
    unsigned short* WoT   = (unsigned short*)(ws);             ws += WBYTES;
    unsigned short* actb  = (unsigned short*)(ws);             ws += PBYTES;
    unsigned short* qproj = (unsigned short*)(ws);             ws += PBYTES;
    unsigned short* kproj = (unsigned short*)(ws);             ws += PBYTES;
    unsigned short* vTb   = (unsigned short*)(ws);             ws += PBYTES;
    unsigned short* ctxb  = (unsigned short*)(ws);             ws += PBYTES;

    const int wn = D_FEAT * D_FEAT;
    dim3 tcg((wn + 255) / 256), tcb(256);
    transpose_cvt_kernel<<<tcg, tcb, 0, stream>>>(Wq, WqT, D_FEAT, D_FEAT);
    transpose_cvt_kernel<<<tcg, tcb, 0, stream>>>(Wk, WkT, D_FEAT, D_FEAT);
    transpose_cvt_kernel<<<tcg, tcb, 0, stream>>>(Wv, WvT, D_FEAT, D_FEAT);
    transpose_cvt_kernel<<<tcg, tcb, 0, stream>>>(Wo, WoT, D_FEAT, D_FEAT);

    const int n8 = MROWS * D_FEAT / 8;
    dim3 cvg((n8 + 255) / 256);
    dim3 gg(MROWS / 128, D_FEAT / 64), gb(128);

    // q -> actb -> qproj ; reuse actb (stream-ordered)
    cvt_bf16_kernel<<<cvg, tcb, 0, stream>>>(q, actb, n8);
    gemm_kernel<<<gg, gb, 0, stream>>>(actb, WqT, bq, qproj, nullptr, nullptr, 0);
    cvt_bf16_kernel<<<cvg, tcb, 0, stream>>>(k, actb, n8);
    gemm_kernel<<<gg, gb, 0, stream>>>(actb, WkT, bk, kproj, nullptr, nullptr, 0);
    cvt_bf16_kernel<<<cvg, tcb, 0, stream>>>(v, actb, n8);
    gemm_kernel<<<gg, gb, 0, stream>>>(actb, WvT, bv, vTb, nullptr, nullptr, 1);

    attention_kernel<<<dim3(BATCH * NHEAD * (SEQ / 16) / 4), gb, 0, stream>>>(
        qproj, kproj, vTb, ctxb);

    gemm_kernel<<<gg, gb, 0, stream>>>(ctxb, WoT, bo, nullptr, v, out, 2);
    layernorm_kernel<<<dim3(MROWS), dim3(256), 0, stream>>>(out, gamma, beta);
}
